// Decoder_19834158973364
// MI455X (gfx1250) — compile-verified, matrix operand order checked by reading, not run on hardware
//
#include <hip/hip_runtime.h>

// ---------------------------------------------------------------------------
// CDNA5 (gfx1250) GNN mesh decoder.
// Heavy GEMMs (edge MLP, node update, decoder stage-1) run on
// v_wmma_f32_16x16x32_bf16. Packed bf16 weights are staged into LDS once per
// block via global_load_async_to_lds_b128 (ASYNCcnt) and consumed with
// ds_load_b128. Scatter-add uses global_atomic_add_f32.
// wave32: one wave owns a 16-row tile x 128 output columns (8 col-tiles).
// ---------------------------------------------------------------------------

typedef __attribute__((ext_vector_type(16))) __bf16 v16bf;
typedef __attribute__((ext_vector_type(8)))  float  v8f;

namespace {
constexpr int Bn  = 4;
constexpr int Nf  = 100000;   // fine nodes
constexpr int Nm  = 25000;    // mid nodes
constexpr int Nc  = 6250;     // coarse nodes
constexpr int E0c = 300000;
constexpr int E1c = 75000;
constexpr int E2c = 18750;
constexpr int Hc  = 128;
constexpr int Lc  = 128;
}

static __device__ __forceinline__ unsigned short f32_bf16(float f) {
  union { float f; unsigned u; } c; c.f = f;
  unsigned u = c.u + 0x7FFFu + ((c.u >> 16) & 1u);   // round-to-nearest-even
  return (unsigned short)(u >> 16);
}

// Two f32 -> packed 2x bf16 (round-half-up) in one v_perm_b32.
static __device__ __forceinline__ unsigned pack_bf16x2(float lo, float hi) {
  union { float f; unsigned u; } a, b;
  a.f = lo; b.f = hi;
  unsigned ra = a.u + 0x8000u;
  unsigned rb = b.u + 0x8000u;
  // result bytes: [0]=ra.b2 [1]=ra.b3 [2]=rb.b2 [3]=rb.b3
  return __builtin_amdgcn_perm(rb, ra, 0x07060302u);
}

union V16U { v16bf v; unsigned u[8]; };

// Load 16 consecutive f32 (64B) and convert to a bf16 A-fragment half-row.
static __device__ __forceinline__ v16bf loadA_bf16(const float* __restrict__ p) {
  const float4* p4 = reinterpret_cast<const float4*>(p);
  float4 a = p4[0], b = p4[1], c = p4[2], d = p4[3];
  V16U u;
  u.u[0] = pack_bf16x2(a.x, a.y);
  u.u[1] = pack_bf16x2(a.z, a.w);
  u.u[2] = pack_bf16x2(b.x, b.y);
  u.u[3] = pack_bf16x2(b.z, b.w);
  u.u[4] = pack_bf16x2(c.x, c.y);
  u.u[5] = pack_bf16x2(c.z, c.w);
  u.u[6] = pack_bf16x2(d.x, d.y);
  u.u[7] = pack_bf16x2(d.z, d.w);
  return u.v;
}

static __device__ __forceinline__ v8f vzero8() {
  v8f v;
#pragma unroll
  for (int i = 0; i < 8; ++i) v[i] = 0.0f;
  return v;
}

// Cooperative async copy of NBYTES (16B-multiple) from global to LDS.
// Issues global_load_async_to_lds_b128; waits on ASYNCcnt itself; caller must barrier.
static __device__ __forceinline__ void async_stage_lds(
    const unsigned short* __restrict__ gsrc, unsigned lds0, int nbytes, int tid) {
  for (int off = tid * 16; off < nbytes; off += 256 * 16) {
    unsigned long long ga = (unsigned long long)((const char*)gsrc + off);
    unsigned la = lds0 + (unsigned)off;
    asm volatile("global_load_async_to_lds_b128 %0, %1, off"
                 :: "v"(la), "v"(ga) : "memory");
  }
  asm volatile("s_wait_asynccnt 0x0" ::: "memory");
}

// ---------------------------------------------------------------------------
// Weight packer: f32 (K x Nout) row-major -> bf16 WMMA B-fragment layout.
// Fragment (kt, ct): 32 lanes x 16 bf16; lane l (l<16): col=ct*16+l, K=kt*32..+15;
// lane l>=16: col=ct*16+(l-16), K=kt*32+16..+31.  One coalesced 32B/lane load.
// ---------------------------------------------------------------------------
__global__ void k_pack_w(const float* __restrict__ W, unsigned short* __restrict__ P,
                         int K, int Nout) {
  int t = blockIdx.x * blockDim.x + threadIdx.x;
  int total = K * Nout;
  if (t >= total) return;
  int k = t / Nout, c = t % Nout;
  int NT = Nout >> 4;
  int kt = k >> 5, kin = k & 31, hi = kin >> 4, j = kin & 15;
  int ct = c >> 4, l = hi * 16 + (c & 15);
  P[(((size_t)(kt * NT + ct) * 32) + l) * 16 + j] = f32_bf16(W[t]);
}

// z:(B,L,Nc) -> zx:(B,Nc,L)   (write-coalesced)
__global__ void k_transpose(const float* __restrict__ z, float* __restrict__ zx) {
  int t = blockIdx.x * blockDim.x + threadIdx.x;
  if (t >= Bn * Nc * Lc) return;
  int l = t % Lc;
  int n = (t / Lc) % Nc;
  int b = t / (Lc * Nc);
  zx[t] = z[((size_t)b * Lc + l) * Nc + n];
}

// out[b, idx[i], cooff + c] = in[b, i, c]   (out pre-zeroed by memset)
__global__ void k_unpool(const float* __restrict__ in, float* __restrict__ out,
                         const int* __restrict__ idx, int nIn, int C,
                         int NnOut, int ostride, int cooff) {
  int t = blockIdx.x * blockDim.x + threadIdx.x;
  int total = Bn * nIn * C;
  if (t >= total) return;
  int c = t % C;
  int i = (t / C) % nIn;
  int b = t / (C * nIn);
  out[((size_t)b * NnOut + idx[i]) * ostride + cooff + c] = in[t];
}

// ---------------------------------------------------------------------------
// Edge MLP: m = relu([x_src | x_dst] @ Wm); scatter-add m into agg[dst].
// One wave = 16 edges x 128 message columns. K = 2*FIN, bf16 WMMA.
// Whole packed Wm staged into LDS asynchronously once per block.
// ---------------------------------------------------------------------------
template <int FIN>
__global__ __launch_bounds__(256) void k_edge_mlp(
    const float* __restrict__ x,          // (B, Nn, FIN)
    const int* __restrict__ edges,        // (2, E): row0=src, row1=dst
    const unsigned short* __restrict__ Wp,// packed bf16 (2*FIN x 128)
    float* __restrict__ agg,              // (B, Nn, 128)
    int Nn, int E) {
  constexpr int KT = (2 * FIN) / 32;
  constexpr int WBYTES = KT * 8 * 32 * 16 * 2;      // KT*8 frags * 1KB
  __shared__ unsigned short wlds[KT * 8 * 32 * 16];

  const int lane = threadIdx.x & 31;
  const int wave = threadIdx.x >> 5;
  const int tile = blockIdx.x * 8 + wave;
  const bool active = (tile * 16 < E);              // wave-uniform

  async_stage_lds(Wp, (unsigned)(uintptr_t)&wlds[0], WBYTES, threadIdx.x);
  __syncthreads();
  if (!active) return;

  const int b  = blockIdx.y;
  const int r  = lane & 15;
  const int hi = lane >> 4;

  int e = tile * 16 + r;
  if (e >= E) e = E - 1;                 // clamp ragged tile (reads only)
  const int sn = edges[e];
  const float* xs = x + ((size_t)b * Nn + sn) * FIN;
  const int dn = edges[E + e];
  const float* xd = x + ((size_t)b * Nn + dn) * FIN;
  const v16bf* Bs = reinterpret_cast<const v16bf*>(wlds);

  v8f acc[8];
#pragma unroll
  for (int ct = 0; ct < 8; ++ct) acc[ct] = vzero8();

  for (int kt = 0; kt < KT; ++kt) {
    const int kk = kt * 32 + hi * 16;
    const float* ap = (kk < FIN) ? (xs + kk) : (xd + (kk - FIN));
    v16bf af = loadA_bf16(ap);
#pragma unroll
    for (int ct = 0; ct < 8; ++ct) {
      v16bf bf = Bs[(kt * 8 + ct) * 32 + lane];
      acc[ct] = __builtin_amdgcn_wmma_f32_16x16x32_bf16(
          false, af, false, bf, (short)0, acc[ct], false, false);
    }
  }

  // relu + scatter-add. D layout: VGPR v -> row v+8*hi, col = ct*16 + r.
  float* aggb = agg + (size_t)b * Nn * Hc;
#pragma unroll
  for (int v = 0; v < 8; ++v) {
    const int ev = tile * 16 + hi * 8 + v;
    if (ev < E) {
      const int d2 = edges[E + ev];
      float* rowp = aggb + (size_t)d2 * Hc + r;
#pragma unroll
      for (int ct = 0; ct < 8; ++ct) {
        float m = acc[ct][v];
        m = m > 0.0f ? m : 0.0f;
        atomicAdd(rowp + ct * 16, m);
      }
    }
  }
}

// ---------------------------------------------------------------------------
// Node update: out = relu([x | agg] @ Wu). Rows flattened over B*Nn.
// Supports strided/offset output (for writing into concat buffers) and
// in-place (out == x): each row is read only by its owning wave, and all
// A-fragment reads precede the stores.
// ---------------------------------------------------------------------------
template <int FIN>
__global__ __launch_bounds__(256) void k_node_update(
    const float* __restrict__ x,          // (Mtot, FIN)
    const float* __restrict__ agg,        // (Mtot, 128)
    const unsigned short* __restrict__ Wp,// packed bf16 ((FIN+128) x 128)
    float* __restrict__ out, int Mtot, int ostride, int cooff) {
  constexpr int KT = (FIN + Hc) / 32;
  constexpr int WBYTES = KT * 8 * 32 * 16 * 2;
  __shared__ unsigned short wlds[KT * 8 * 32 * 16];

  const int lane = threadIdx.x & 31;
  const int wave = threadIdx.x >> 5;
  const int tile = blockIdx.x * 8 + wave;
  const bool active = (tile * 16 < Mtot);           // wave-uniform

  async_stage_lds(Wp, (unsigned)(uintptr_t)&wlds[0], WBYTES, threadIdx.x);
  __syncthreads();
  if (!active) return;

  const int r  = lane & 15;
  const int hi = lane >> 4;

  int row = tile * 16 + r;
  if (row >= Mtot) row = Mtot - 1;
  const float* xr = x + (size_t)row * FIN;
  const float* ar = agg + (size_t)row * Hc;
  const v16bf* Bs = reinterpret_cast<const v16bf*>(wlds);

  v8f acc[8];
#pragma unroll
  for (int ct = 0; ct < 8; ++ct) acc[ct] = vzero8();

  for (int kt = 0; kt < KT; ++kt) {
    const int kk = kt * 32 + hi * 16;
    const float* ap = (kk < FIN) ? (xr + kk) : (ar + (kk - FIN));
    v16bf af = loadA_bf16(ap);
#pragma unroll
    for (int ct = 0; ct < 8; ++ct) {
      v16bf bf = Bs[(kt * 8 + ct) * 32 + lane];
      acc[ct] = __builtin_amdgcn_wmma_f32_16x16x32_bf16(
          false, af, false, bf, (short)0, acc[ct], false, false);
    }
  }

#pragma unroll
  for (int v = 0; v < 8; ++v) {
    const int rv = tile * 16 + hi * 8 + v;
    if (rv < Mtot) {
      float* op = out + (size_t)rv * ostride + cooff + r;
#pragma unroll
      for (int ct = 0; ct < 8; ++ct) {
        float m = acc[ct][v];
        op[ct * 16] = m > 0.0f ? m : 0.0f;
      }
    }
  }
}

// ---------------------------------------------------------------------------
// Decoder: h = leaky_relu(x @ Wd1 + bd1)  (WMMA, 128->64)
//          o = h @ Wd2 + bd2              (scalar via LDS transpose, 64->3)
//          LayerNorm over the 3 outputs.
// ---------------------------------------------------------------------------
__global__ __launch_bounds__(256) void k_decoder(
    const float* __restrict__ x,            // (Mtot, 128)
    const unsigned short* __restrict__ Wd1p,// packed bf16 (128 x 64)
    const float* __restrict__ bd1, const float* __restrict__ Wd2,
    const float* __restrict__ bd2, const float* __restrict__ gamma,
    const float* __restrict__ beta, float* __restrict__ out, int Mtot) {
  __shared__ unsigned short wlds[4 * 4 * 32 * 16];   // 16KB packed Wd1
  __shared__ float hs[8 * 16 * 65];
  const int lane = threadIdx.x & 31;
  const int wave = threadIdx.x >> 5;
  const int tile = blockIdx.x * 8 + wave;
  const int r  = lane & 15;
  const int hi = lane >> 4;
  const bool active = (tile * 16 < Mtot);   // wave-uniform

  async_stage_lds(Wd1p, (unsigned)(uintptr_t)&wlds[0], 4 * 4 * 32 * 16 * 2,
                  threadIdx.x);
  __syncthreads();

  if (active) {
    int row = tile * 16 + r;
    if (row >= Mtot) row = Mtot - 1;
    const float* xr = x + (size_t)row * Hc;
    const v16bf* Bs = reinterpret_cast<const v16bf*>(wlds);

    v8f acc[4];
#pragma unroll
    for (int ct = 0; ct < 4; ++ct) acc[ct] = vzero8();

#pragma unroll
    for (int kt = 0; kt < 4; ++kt) {
      v16bf af = loadA_bf16(xr + kt * 32 + hi * 16);
#pragma unroll
      for (int ct = 0; ct < 4; ++ct) {
        v16bf bf = Bs[(kt * 4 + ct) * 32 + lane];
        acc[ct] = __builtin_amdgcn_wmma_f32_16x16x32_bf16(
            false, af, false, bf, (short)0, acc[ct], false, false);
      }
    }

#pragma unroll
    for (int ct = 0; ct < 4; ++ct) {
      const int c = ct * 16 + r;
      const float bias = bd1[c];
#pragma unroll
      for (int v = 0; v < 8; ++v) {
        float h = acc[ct][v] + bias;
        h = h > 0.0f ? h : 0.01f * h;     // leaky_relu(0.01)
        hs[(wave * 16 + hi * 8 + v) * 65 + c] = h;
      }
    }
  }
  __syncthreads();

  if (active && lane < 16) {
    const int row = tile * 16 + lane;
    if (row < Mtot) {
      const float* hr = &hs[(wave * 16 + lane) * 65];
      float o[3];
#pragma unroll
      for (int j = 0; j < 3; ++j) o[j] = bd2[j];
      for (int c = 0; c < 64; ++c) {
        const float h = hr[c];
#pragma unroll
        for (int j = 0; j < 3; ++j) o[j] += h * Wd2[c * 3 + j];
      }
      const float mu = (o[0] + o[1] + o[2]) * (1.0f / 3.0f);
      float var = 0.0f;
#pragma unroll
      for (int j = 0; j < 3; ++j) { float d = o[j] - mu; var += d * d; }
      var *= (1.0f / 3.0f);
      const float inv = rsqrtf(var + 1e-5f);
#pragma unroll
      for (int j = 0; j < 3; ++j)
        out[(size_t)row * 3 + j] = (o[j] - mu) * inv * gamma[j] + beta[j];
    }
  }
}

// ---------------------------------------------------------------------------
// Host launcher
// ---------------------------------------------------------------------------
extern "C" void kernel_launch(void* const* d_in, const int* in_sizes, int n_in,
                              void* d_out, int out_size, void* d_ws, size_t ws_size,
                              hipStream_t stream) {
  (void)in_sizes; (void)n_in; (void)out_size; (void)ws_size;

  const float* z   = (const float*)d_in[0];
  const int*   g0  = (const int*)d_in[1];
  const int*   g1  = (const int*)d_in[2];
  const int*   g2  = (const int*)d_in[3];
  const int*   id0 = (const int*)d_in[4];
  const int*   id1 = (const int*)d_in[5];
  const float* Wm0 = (const float*)d_in[6];
  const float* Wu0 = (const float*)d_in[7];
  const float* WmR = (const float*)d_in[8];
  const float* WuR = (const float*)d_in[9];
  const float* Wm1 = (const float*)d_in[10];
  const float* Wu1 = (const float*)d_in[11];
  const float* WmF = (const float*)d_in[12];
  const float* WuF = (const float*)d_in[13];
  const float* Wd1 = (const float*)d_in[14];
  const float* bd1 = (const float*)d_in[15];
  const float* Wd2 = (const float*)d_in[16];
  const float* bd2 = (const float*)d_in[17];
  const float* gam = (const float*)d_in[18];
  const float* bet = (const float*)d_in[19];

  char* ws = (char*)d_ws;
  size_t cur = 0;
  auto take = [&](size_t bytes) -> char* {
    char* p = ws + cur;
    cur = (cur + bytes + 255) & ~(size_t)255;
    return p;
  };

  // packed bf16 weights
  unsigned short* WmRp = (unsigned short*)take((size_t)256 * 128 * 2);
  unsigned short* WuRp = (unsigned short*)take((size_t)256 * 128 * 2);
  unsigned short* Wm0p = (unsigned short*)take((size_t)256 * 128 * 2);
  unsigned short* Wu0p = (unsigned short*)take((size_t)256 * 128 * 2);
  unsigned short* Wm1p = (unsigned short*)take((size_t)512 * 128 * 2);
  unsigned short* Wu1p = (unsigned short*)take((size_t)384 * 128 * 2);
  unsigned short* WmFp = (unsigned short*)take((size_t)256 * 128 * 2);
  unsigned short* WuFp = (unsigned short*)take((size_t)256 * 128 * 2);
  unsigned short* Wd1p = (unsigned short*)take((size_t)128 * 64 * 2);

  // activation / scratch buffers (f32)
  float* Zx   = (float*)take((size_t)Bn * Nc * Lc * 4);       // (B,Nc,L)
  float* U    = (float*)take((size_t)Bn * Nm * Lc * 4);       // unpooled z / later x1
  float* AGG  = (float*)take((size_t)Bn * Nm * Hc * 4);       // mid/coarse aggregates
  float* X2   = (float*)take((size_t)Bn * Nc * Hc * 4);       // coarse _mp output
  float* CAT  = (float*)take((size_t)Bn * Nm * 2 * Hc * 4);   // concat(x_up, res)
  float* X0   = (float*)take((size_t)Bn * Nf * Hc * 4);       // fine features
  float* AGGF = (float*)take((size_t)Bn * Nf * Hc * 4);       // fine aggregates

  auto packW = [&](const float* W, unsigned short* P, int K, int Nw) {
    int total = K * Nw;
    k_pack_w<<<dim3((total + 255) / 256), dim3(256), 0, stream>>>(W, P, K, Nw);
  };
  packW(WmR, WmRp, 256, 128);
  packW(WuR, WuRp, 256, 128);
  packW(Wm0, Wm0p, 256, 128);
  packW(Wu0, Wu0p, 256, 128);
  packW(Wm1, Wm1p, 512, 128);
  packW(Wu1, Wu1p, 384, 128);
  packW(WmF, WmFp, 256, 128);
  packW(WuF, WuFp, 256, 128);
  packW(Wd1, Wd1p, 128, 64);

  // z_x = swapaxes(z)
  {
    int total = Bn * Nc * Lc;
    k_transpose<<<dim3((total + 255) / 256), dim3(256), 0, stream>>>(z, Zx);
  }

  // zero buffers that receive sparse writes
  (void)hipMemsetAsync(U,   0, (size_t)Bn * Nm * Lc * 4, stream);
  (void)hipMemsetAsync(CAT, 0, (size_t)Bn * Nm * 2 * Hc * 4, stream);

  // x1a = unpool(z_x, Nm, id1)
  {
    int total = Bn * Nc * Lc;
    k_unpool<<<dim3((total + 255) / 256), dim3(256), 0, stream>>>(
        Zx, U, id1, Nc, Lc, Nm, Lc, 0);
  }

  // res = _mp(x1a, g1, WmR, WuR)  -> CAT[:, :, 128:256]
  (void)hipMemsetAsync(AGG, 0, (size_t)Bn * Nm * Hc * 4, stream);
  {
    int tiles = (E1c + 15) / 16;
    dim3 g((tiles + 7) / 8, Bn);
    k_edge_mlp<128><<<g, dim3(256), 0, stream>>>(U, g1, WmRp, AGG, Nm, E1c);
  }
  {
    int M = Bn * Nm, tiles = (M + 15) / 16;
    k_node_update<128><<<dim3((tiles + 7) / 8), dim3(256), 0, stream>>>(
        U, AGG, WuRp, CAT, M, 256, 128);
  }

  // x2 = _mp(z_x, g2, Wm0, Wu0)
  (void)hipMemsetAsync(AGG, 0, (size_t)Bn * Nc * Hc * 4, stream);
  {
    int tiles = (E2c + 15) / 16;
    dim3 g((tiles + 7) / 8, Bn);
    k_edge_mlp<128><<<g, dim3(256), 0, stream>>>(Zx, g2, Wm0p, AGG, Nc, E2c);
  }
  {
    int M = Bn * Nc, tiles = (M + 15) / 16;
    k_node_update<128><<<dim3((tiles + 7) / 8), dim3(256), 0, stream>>>(
        Zx, AGG, Wu0p, X2, M, 128, 0);
  }

  // unpool(x2) -> CAT[:, :, 0:128]
  {
    int total = Bn * Nc * Hc;
    k_unpool<<<dim3((total + 255) / 256), dim3(256), 0, stream>>>(
        X2, CAT, id1, Nc, Hc, Nm, 256, 0);
  }

  // x1 = _mp(CAT, g1, Wm1, Wu1)  (Fin=256) -> U
  (void)hipMemsetAsync(AGG, 0, (size_t)Bn * Nm * Hc * 4, stream);
  {
    int tiles = (E1c + 15) / 16;
    dim3 g((tiles + 7) / 8, Bn);
    k_edge_mlp<256><<<g, dim3(256), 0, stream>>>(CAT, g1, Wm1p, AGG, Nm, E1c);
  }
  {
    int M = Bn * Nm, tiles = (M + 15) / 16;
    k_node_update<256><<<dim3((tiles + 7) / 8), dim3(256), 0, stream>>>(
        CAT, AGG, Wu1p, U, M, 128, 0);
  }

  // x0 = unpool(x1, Nf, id0)
  (void)hipMemsetAsync(X0, 0, (size_t)Bn * Nf * Hc * 4, stream);
  {
    int total = Bn * Nm * Hc;
    k_unpool<<<dim3((total + 255) / 256), dim3(256), 0, stream>>>(
        U, X0, id0, Nm, Hc, Nf, 128, 0);
  }

  // xF = _mp(x0, g0, WmF, WuF)  (in-place into X0)
  (void)hipMemsetAsync(AGGF, 0, (size_t)Bn * Nf * Hc * 4, stream);
  {
    int tiles = (E0c + 15) / 16;
    dim3 g((tiles + 7) / 8, Bn);
    k_edge_mlp<128><<<g, dim3(256), 0, stream>>>(X0, g0, WmFp, AGGF, Nf, E0c);
  }
  {
    int M = Bn * Nf, tiles = (M + 15) / 16;
    k_node_update<128><<<dim3((tiles + 7) / 8), dim3(256), 0, stream>>>(
        X0, AGGF, WuFp, X0, M, 128, 0);
  }

  // decoder + layernorm -> d_out
  {
    int M = Bn * Nf, tiles = (M + 15) / 16;
    k_decoder<<<dim3((tiles + 7) / 8), dim3(256), 0, stream>>>(
        X0, Wd1p, bd1, Wd2, bd2, gam, bet, (float*)d_out, M);
  }
}
